// ChunkedParallelmLSTMBlock_70428873720000
// MI455X (gfx1250) — compile-verified, matrix operand order checked by reading, not run on hardware
//
#include <hip/hip_runtime.h>
#include <cstdint>

// ---------------- problem constants ----------------
#define D_IN  768
#define NHH   8
#define HD    96
#define HIDN  768
#define UPD   1536
#define FUSED 2320
#define SEQ   4096
#define BATCH 2
#define NTOK  (BATCH*SEQ)   // 8192
#define CSZ   64
#define NC    (SEQ/CSZ)     // 64
#define CAPV  15.0f

// ---------------- WMMA types ----------------
typedef __attribute__((ext_vector_type(16))) __bf16 bf16x16;
typedef __attribute__((ext_vector_type(8)))  float  floatx8;

union Frag { bf16x16 v; uint32_t u[8]; uint4 q[2]; };
union Acc  { floatx8 v; float f[8]; };

__device__ __forceinline__ floatx8 wmma_bf16(bf16x16 a, bf16x16 b, floatx8 c) {
  // 8 args: (neg_a, A, neg_b, B, c_mod, C, reuse_a, reuse_b)
  return __builtin_amdgcn_wmma_f32_16x16x32_bf16(false, a, false, b, (short)0, c, false, false);
}

__device__ __forceinline__ uint16_t f2bf(float f) {
  uint32_t u = __float_as_uint(f);
  u += 0x7FFFu + ((u >> 16) & 1u);           // round-to-nearest-even
  return (uint16_t)(u >> 16);
}
__device__ __forceinline__ uint32_t packbf(float a, float b) {
  return (uint32_t)f2bf(a) | ((uint32_t)f2bf(b) << 16);
}

// ---- fragment-ready LDS layout helpers ----
// A 16x32-bf16 fragment for lane L=(hi*16+m): u[0..3] = K-pairs hi*4+0..3,
// u[4..7] = K-pairs 8+hi*4+0..3.  We store operand tiles as [tile][lane][8 dwords]
// so a fragment load is exactly two ds_load_b128 into the WMMA source VGPRs.
__device__ __forceinline__ int frag_slot(int kk, int& hi) {
  hi = (kk >> 2) & 1;
  return (kk & 3) + ((kk >> 3) << 2);
}
// element (row, kp) of an A-side operand with `ksteps` K-steps of 16 pairs
__device__ __forceinline__ int fragA_idx(int row, int kp, int ksteps) {
  int hi; int slot = frag_slot(kp & 15, hi);
  return (((row >> 4) * ksteps + (kp >> 4)) * 32 + hi * 16 + (row & 15)) * 8 + slot;
}
// element (kp, col) of a B-side operand with `ksteps` K-steps of 16 pairs
__device__ __forceinline__ int fragB_idx(int kp, int col, int ksteps) {
  int hi; int slot = frag_slot(kp & 15, hi);
  return (((col >> 4) * ksteps + (kp >> 4)) * 32 + hi * 16 + (col & 15)) * 8 + slot;
}
__device__ __forceinline__ bf16x16 load_frag_ready(const uint32_t* base, int tile, int lane) {
  Frag f;
  const uint4* p = (const uint4*)(base + (size_t)(tile * 32 + lane) * 8);
  f.q[0] = p[0];           // ds_load_b128
  f.q[1] = p[1];           // ds_load_b128
  return f.v;
}

// ---------------- weight pack: fp32 [K][N] -> bf16-pair uint [K/2][N] ----------------
__global__ void pack_weight_kernel(const float* __restrict__ W, uint32_t* __restrict__ Wp, int K, int N) {
  int idx = blockIdx.x * 256 + threadIdx.x;
  int total = (K >> 1) * N;
  if (idx >= total) return;
  int kp = idx / N, n = idx % N;
  Wp[idx] = packbf(W[(size_t)(2 * kp) * N + n], W[(size_t)(2 * kp + 1) * N + n]);
}

// ---------------- input layernorm -> bf16 ----------------
__global__ void __launch_bounds__(256) ln_in_kernel(const float* __restrict__ x,
    const float* __restrict__ w, const float* __restrict__ b, uint16_t* __restrict__ xnb) {
  __shared__ float red[256];
  int tok = blockIdx.x, tid = threadIdx.x;
  const float* xr = x + (size_t)tok * D_IN;
  float vals[3]; float s = 0.f;
#pragma unroll
  for (int i = 0; i < 3; ++i) { vals[i] = xr[tid + i * 256]; s += vals[i]; }
  red[tid] = s; __syncthreads();
  for (int o = 128; o > 0; o >>= 1) { if (tid < o) red[tid] += red[tid + o]; __syncthreads(); }
  float mean = red[0] / (float)D_IN; __syncthreads();
  s = 0.f;
#pragma unroll
  for (int i = 0; i < 3; ++i) { float d = vals[i] - mean; s += d * d; }
  red[tid] = s; __syncthreads();
  for (int o = 128; o > 0; o >>= 1) { if (tid < o) red[tid] += red[tid + o]; __syncthreads(); }
  float rstd = rsqrtf(red[0] / (float)D_IN + 1e-6f);
#pragma unroll
  for (int i = 0; i < 3; ++i) {
    int c = tid + i * 256;
    xnb[(size_t)tok * D_IN + c] = f2bf((vals[i] - mean) * rstd * w[c] + b[c]);
  }
}

// ---------------- WMMA GEMM: C[M,N] = bf16(A[M,K]) @ bf16(W[K,N]) + bias (+epi) ----------------
// EPI: 0 = bias, 1 = sigmoid(bias+..), 2 = bias + residual
template <int EPI>
__global__ void __launch_bounds__(256) gemm_bf16_kernel(
    const uint16_t* __restrict__ A, const uint32_t* __restrict__ Wp,
    const float* __restrict__ bias, const float* __restrict__ resid,
    float* __restrict__ C, int M, int N, int K) {
  __shared__ uint32_t As_fr[8 * 32 * 8];   // 8 M-tiles, fragment-ready
  __shared__ uint32_t Bs_fr[8 * 32 * 8];   // 8 N-tiles, fragment-ready
  int tid = threadIdx.x;
  int lane = tid & 31, wid = tid >> 5;
  int l15 = lane & 15;
  int mb = blockIdx.y * 128, nb = blockIdx.x * 128;
  int wm = (wid & 1) * 64, wn = (wid >> 1) * 32;
  int K2 = K >> 1;
  const uint32_t* A32 = (const uint32_t*)A;

  floatx8 zero = {0.f, 0.f, 0.f, 0.f, 0.f, 0.f, 0.f, 0.f};
  floatx8 acc[4][2];
#pragma unroll
  for (int i = 0; i < 4; ++i) { acc[i][0] = zero; acc[i][1] = zero; }

  // per-thread staging roles (constant across K loop)
  int ar = tid >> 1, ahalf = tid & 1;              // A: 2 threads/row, 8 dwords each
  int amt = ar >> 4, al15 = ar & 15;
  int bkp = tid >> 4, bn0 = (tid & 15) * 8;        // B: 16 threads/kp-row, 8 cols each
  int bhi; int bslot = frag_slot(bkp, bhi);

  for (int kt = 0; kt < K; kt += 32) {
    int kp_base = kt >> 1;
    {   // stage A: two 16B global reads -> two 16B fragment-ready LDS writes
      const uint4* src = (const uint4*)(A32 + (size_t)(mb + ar) * K2 + kp_base + ahalf * 8);
      uint4 a0 = src[0], a1 = src[1];
      uint4* dst0 = (uint4*)(As_fr + (size_t)(amt * 32 + al15) * 8);        // hi=0 lane image
      uint4* dst1 = (uint4*)(As_fr + (size_t)(amt * 32 + 16 + al15) * 8);   // hi=1 lane image
      dst0[ahalf] = a0;
      dst1[ahalf] = a1;
    }
    {   // stage B: coalesced global reads, swizzled fragment-ready scatter
      const uint32_t* src = Wp + (size_t)(kp_base + bkp) * N + nb + bn0;
#pragma unroll
      for (int u = 0; u < 8; ++u) {
        int n = bn0 + u;
        uint32_t val = (nb + n < N) ? src[u] : 0u;
        Bs_fr[((n >> 4) * 32 + bhi * 16 + (n & 15)) * 8 + bslot] = val;
      }
    }
    if (kt + 32 < K) {   // global_prefetch_b8 for next K tile
      __builtin_prefetch(A32 + (size_t)(mb + ar) * K2 + kp_base + 16 + ahalf * 8, 0, 1);
      __builtin_prefetch(Wp + (size_t)(kp_base + 16 + bkp) * N + nb + bn0, 0, 1);
    }
    __syncthreads();
    bf16x16 bfrag[2];
#pragma unroll
    for (int nt = 0; nt < 2; ++nt) bfrag[nt] = load_frag_ready(Bs_fr, (wn >> 4) + nt, lane);
#pragma unroll
    for (int mt = 0; mt < 4; ++mt) {
      bf16x16 afrag = load_frag_ready(As_fr, (wm >> 4) + mt, lane);
      acc[mt][0] = wmma_bf16(afrag, bfrag[0], acc[mt][0]);
      acc[mt][1] = wmma_bf16(afrag, bfrag[1], acc[mt][1]);
    }
    __syncthreads();
  }
  int hi = lane >> 4;
#pragma unroll
  for (int mt = 0; mt < 4; ++mt)
#pragma unroll
    for (int nt = 0; nt < 2; ++nt) {
      int n = nb + wn + nt * 16 + l15;
      if (n >= N) continue;
      Acc a; a.v = acc[mt][nt];
      float bval = bias[n];
#pragma unroll
      for (int r = 0; r < 8; ++r) {
        int row = mb + wm + mt * 16 + r + 8 * hi;
        float v = a.f[r] + bval;
        if (EPI == 1) v = 1.0f / (1.0f + __expf(-v));
        if (EPI == 2) v += resid[(size_t)row * N + n];
        C[(size_t)row * N + n] = v;
      }
    }
}

// ---------------- causal conv (KER=4 shared taps) + SiLU, also bf16 copies ----------------
__global__ void __launch_bounds__(256) conv_silu_kernel(const float* __restrict__ xt,
    const float* __restrict__ cw, const float* __restrict__ cb,
    uint16_t* __restrict__ xcb, uint16_t* __restrict__ xtb) {
  int tok = blockIdx.x, tid = threadIdx.x;
  int s = tok % SEQ;
  float w0 = cw[0], w1 = cw[1], w2 = cw[2], w3 = cw[3], bias = cb[0];
#pragma unroll
  for (int i = 0; i < UPD / 256; ++i) {
    int ch = tid + i * 256;
    float cur = xt[(size_t)tok * UPD + ch];
    float acc = bias + w3 * cur;
    if (s >= 1) acc += w2 * xt[(size_t)(tok - 1) * UPD + ch];
    if (s >= 2) acc += w1 * xt[(size_t)(tok - 2) * UPD + ch];
    if (s >= 3) acc += w0 * xt[(size_t)(tok - 3) * UPD + ch];
    float sv = acc / (1.0f + __expf(-acc));
    xcb[(size_t)tok * UPD + ch] = f2bf(sv);
    xtb[(size_t)tok * UPD + ch] = f2bf(cur);
  }
}

// ---------------- gates helper ----------------
__device__ __forceinline__ void compute_gate(const float* __restrict__ proj, int t, int nh,
                                             float& ig, float& lg) {
  float ip = proj[(size_t)t * FUSED + nh];
  float fp = proj[(size_t)t * FUSED + NHH + nh];
  ip = CAPV * tanhf(ip * (1.0f / CAPV));
  fp = CAPV * tanhf(fp * (1.0f / CAPV));
  float mx = fmaxf(ip, fp);
  ig = __expf(ip - mx);
  lg = __logf(__expf(fp - mx) + 1e-8f);
}

// ---------------- per-chunk state contributions (fp32 for recurrent path) ----------------
__global__ void __launch_bounds__(256) chunk_contrib_kernel(const float* __restrict__ proj,
    float* __restrict__ Ccon, float* __restrict__ ncon) {
  __shared__ float kf[CSZ * HD];
  __shared__ float vf[CSZ * HD];
  __shared__ float wC[CSZ];
  __shared__ float lf[CSZ];
  int gid = blockIdx.x;
  int c = gid & (NC - 1), nh = (gid >> 6) & (NHH - 1), b = gid >> 9;
  int tid = threadIdx.x;
  int tbase = b * SEQ + c * CSZ;
  if (tid < CSZ) {
    float igv, lg; compute_gate(proj, tbase + tid, nh, igv, lg);
    wC[tid] = igv; lf[tid] = lg;
  }
  __syncthreads();
  if (tid == 0) { float r = 0.f; for (int l = 0; l < CSZ; ++l) { r += lf[l]; lf[l] = r; } }
  __syncthreads();
  if (tid < CSZ) wC[tid] = __expf(lf[tid] - lf[CSZ - 1]) * wC[tid];
  const float rs = 0.1020620726f;  // 1/sqrt(96)
  for (int idx = tid; idx < CSZ * HD; idx += 256) {
    int l = idx / HD, d = idx % HD;
    size_t t = (size_t)(tbase + l);
    kf[idx] = proj[t * FUSED + 2 * NHH + HIDN + nh * HD + d] * rs;
    vf[idx] = proj[t * FUSED + 2 * NHH + 2 * HIDN + nh * HD + d];
  }
  __syncthreads();
  size_t obase = (size_t)gid * (HD * HD);
  for (int idx = tid; idx < HD * HD; idx += 256) {
    int d = idx / HD, e = idx % HD;
    float s = 0.f;
    for (int l = 0; l < CSZ; ++l) s += wC[l] * vf[l * HD + d] * kf[l * HD + e];
    Ccon[obase + idx] = s;
  }
  if (tid < HD) {
    float s = 0.f;
    for (int l = 0; l < CSZ; ++l) s += wC[l] * kf[l * HD + tid];
    ncon[(size_t)gid * HD + tid] = s;
  }
}

// ---------------- exclusive scan of chunk states over NC ----------------
__global__ void __launch_bounds__(256) scan_kernel(const float* __restrict__ Ccon,
    const float* __restrict__ ncon, float* __restrict__ Cini, float* __restrict__ nini) {
  int bh = blockIdx.x, tid = threadIdx.x;
  float acc[36];
#pragma unroll
  for (int j = 0; j < 36; ++j) acc[j] = 0.f;
  float an = 0.f;
  for (int c = 0; c < NC; ++c) {
    size_t base = ((size_t)bh * NC + c) * (HD * HD);
#pragma unroll
    for (int j = 0; j < 36; ++j) {
      int idx = tid + j * 256;
      Cini[base + idx] = acc[j];
      acc[j] += Ccon[base + idx];
    }
    if (tid < HD) {
      size_t nb_ = ((size_t)bh * NC + c) * HD + tid;
      nini[nb_] = an;
      an += ncon[nb_];
    }
  }
}

// ---------------- WMMA chunk attention (dynamic-LDS arena, ~168 KB of 320 KB WGP LDS) ----------------
// word offsets (all 16B aligned; packed operand arenas are fragment-ready)
#define OQF    0        // f32 q           [64][96]
#define OQP    6144     // u32 q frags     (A, ksteps=3)  3072
#define OKST   9216     // u32 k^T frags   (B, ksteps=3)  3072
#define OKP2   12288    // u32 k frags     (B, ksteps=2)  3072
#define OVP    15360    // u32 v frags     (B, ksteps=2)  3072
#define OCP    18432    // u32 C^T frags   (B, ksteps=3)  4608
#define OSTG   23040    // f32 stage       [64][96]   (k, then v, then n_intra)
#define OATTN  29184    // f32 attn        [64][66]
#define OWF    33408    // f32 decay w     [64][66]
#define OAWP   37632    // u32 (attn*w)    (A, ksteps=2)  2048
#define OWP    39680    // u32 w frags     (A, ksteps=2)  2048
#define OLF    41728
#define OIG    41792
#define ONIN   41856
#define ODEN   41952
#define SMEM_WORDS 42016

__global__ void __launch_bounds__(256) chunk_out_kernel(const float* __restrict__ proj,
    const float* __restrict__ Cini, const float* __restrict__ nini, float* __restrict__ hbuf) {
  extern __shared__ float sm[];
  uint32_t* smu = (uint32_t*)sm;
  float*    qf    = sm  + OQF;
  uint32_t* qp    = smu + OQP;
  uint32_t* ksT   = smu + OKST;
  uint32_t* kp2   = smu + OKP2;
  uint32_t* vp    = smu + OVP;
  uint32_t* Cp    = smu + OCP;
  float*    stage = sm  + OSTG;
  float*    attn  = sm  + OATTN;
  float*    wfb   = sm  + OWF;
  uint32_t* awp   = smu + OAWP;
  uint32_t* wp_   = smu + OWP;
  float*    lf    = sm  + OLF;
  float*    ig    = sm  + OIG;
  float*    nin   = sm  + ONIN;
  float*    den   = sm  + ODEN;

  int gid = blockIdx.x;
  int c = gid & (NC - 1), nh = (gid >> 6) & (NHH - 1), b = gid >> 9;
  int tid = threadIdx.x;
  int lane = tid & 31, wid = tid >> 5;
  int l15 = lane & 15, hi = lane >> 4;
  int tbase = b * SEQ + c * CSZ;
  size_t cbase = (size_t)gid * (HD * HD);
  size_t nbase = (size_t)gid * HD;

  // gates + in-chunk log-forget prefix sum
  if (tid < CSZ) {
    float igv, lg; compute_gate(proj, tbase + tid, nh, igv, lg);
    ig[tid] = igv; lf[tid] = lg;
  }
  __syncthreads();
  if (tid == 0) { float r = 0.f; for (int l = 0; l < CSZ; ++l) { r += lf[l]; lf[l] = r; } }

  // load q (fp32 kept for denom) and scaled k into stage
  const float rs = 0.1020620726f;
  for (int idx = tid; idx < CSZ * HD; idx += 256) {
    int l = idx / HD, d = idx % HD;
    size_t t = (size_t)(tbase + l);
    qf[idx]    = proj[t * FUSED + 2 * NHH + nh * HD + d];
    stage[idx] = proj[t * FUSED + 2 * NHH + HIDN + nh * HD + d] * rs;
  }
  __syncthreads();
  for (int idx = tid; idx < CSZ * (HD / 2); idx += 256) {        // q frags (A, 3 ksteps)
    int i = idx / (HD / 2), kp = idx % (HD / 2);
    qp[fragA_idx(i, kp, 3)] = packbf(qf[i * HD + 2 * kp], qf[i * HD + 2 * kp + 1]);
  }
  for (int idx = tid; idx < (HD / 2) * CSZ; idx += 256) {        // k^T frags (B, 3 ksteps)
    int kp = idx / CSZ, j = idx % CSZ;
    ksT[fragB_idx(kp, j, 3)] = packbf(stage[j * HD + 2 * kp], stage[j * HD + 2 * kp + 1]);
  }
  for (int idx = tid; idx < (CSZ / 2) * HD; idx += 256) {        // k frags (B, 2 ksteps)
    int lp = idx / HD, d = idx % HD;
    kp2[fragB_idx(lp, d, 2)] = packbf(stage[(2 * lp) * HD + d], stage[(2 * lp + 1) * HD + d]);
  }
  for (int idx = tid; idx < (HD / 2) * HD; idx += 256) {         // C_init^T frags (B, 3 ksteps)
    int ep = idx / HD, d = idx % HD;
    Cp[fragB_idx(ep, d, 3)] =
        packbf(Cini[cbase + (size_t)d * HD + 2 * ep], Cini[cbase + (size_t)d * HD + 2 * ep + 1]);
  }
  if (tid < HD) nin[tid] = nini[nbase + tid];
  __syncthreads();
  for (int idx = tid; idx < CSZ * HD; idx += 256) {              // v into stage
    int l = idx / HD, d = idx % HD;
    size_t t = (size_t)(tbase + l);
    stage[idx] = proj[t * FUSED + 2 * NHH + 2 * HIDN + nh * HD + d];
  }
  __syncthreads();
  for (int idx = tid; idx < (CSZ / 2) * HD; idx += 256) {        // v frags (B, 2 ksteps)
    int lp = idx / HD, d = idx % HD;
    vp[fragB_idx(lp, d, 2)] = packbf(stage[(2 * lp) * HD + d], stage[(2 * lp + 1) * HD + d]);
  }
  __syncthreads();

  // scores = q @ k^T  (M=64,N=64,K=96): 16 tiles, 2 per wave
  floatx8 zero = {0.f, 0.f, 0.f, 0.f, 0.f, 0.f, 0.f, 0.f};
#pragma unroll
  for (int q2 = 0; q2 < 2; ++q2) {
    int t = wid * 2 + q2;
    int mt = t >> 2, nt = t & 3;
    floatx8 a = zero;
#pragma unroll
    for (int ks = 0; ks < 3; ++ks) {
      bf16x16 af = load_frag_ready(qp, mt * 3 + ks, lane);
      bf16x16 bf = load_frag_ready(ksT, nt * 3 + ks, lane);
      a = wmma_bf16(af, bf, a);
    }
    Acc aa; aa.v = a;
#pragma unroll
    for (int r = 0; r < 8; ++r) attn[(mt * 16 + r + 8 * hi) * 66 + nt * 16 + l15] = aa.f[r];
  }
  __syncthreads();

  // strictly-causal softmax + decay weights (row owners)
  if (tid < CSZ) {
    int i = tid;
    if (i == 0) {
      for (int j = 0; j < CSZ; ++j) { attn[j] = 0.f; wfb[j] = 0.f; }
    } else {
      float m = -3.0e38f;
      for (int j = 0; j < i; ++j) m = fmaxf(m, attn[i * 66 + j]);
      float ssum = 0.f;
      for (int j = 0; j < i; ++j) { float e = __expf(attn[i * 66 + j] - m); attn[i * 66 + j] = e; ssum += e; }
      float inv = 1.0f / ssum;
      float lfi = lf[i];
      for (int j = 0; j < CSZ; ++j) {
        if (j < i) {
          float w = ig[j] * __expf(lfi - lf[j]);
          wfb[i * 66 + j] = w;
          attn[i * 66 + j] = attn[i * 66 + j] * inv * w;
        } else { wfb[i * 66 + j] = 0.f; attn[i * 66 + j] = 0.f; }
      }
    }
  }
  __syncthreads();
  for (int idx = tid; idx < CSZ * (CSZ / 2); idx += 256) {       // aw, w frags (A, 2 ksteps)
    int i = idx / (CSZ / 2), jp = idx % (CSZ / 2);
    int di = fragA_idx(i, jp, 2);
    awp[di] = packbf(attn[i * 66 + 2 * jp], attn[i * 66 + 2 * jp + 1]);
    wp_[di] = packbf(wfb[i * 66 + 2 * jp], wfb[i * 66 + 2 * jp + 1]);
  }
  __syncthreads();

  // h = (attn*w)@v + q@C_init^T ; n_intra = w@k   (M=64,N=96): 24 tiles, 3 per wave
  floatx8 hacc[3];
#pragma unroll
  for (int q3 = 0; q3 < 3; ++q3) {
    int t = wid * 3 + q3;
    int mt = t / 6, nt = t % 6;
    floatx8 ha = zero, na = zero;
#pragma unroll
    for (int ks = 0; ks < 2; ++ks) {
      bf16x16 aaw = load_frag_ready(awp, mt * 2 + ks, lane);
      bf16x16 bv  = load_frag_ready(vp, nt * 2 + ks, lane);
      ha = wmma_bf16(aaw, bv, ha);
      bf16x16 aw2 = load_frag_ready(wp_, mt * 2 + ks, lane);
      bf16x16 bk  = load_frag_ready(kp2, nt * 2 + ks, lane);
      na = wmma_bf16(aw2, bk, na);
    }
#pragma unroll
    for (int ks = 0; ks < 3; ++ks) {
      bf16x16 aq = load_frag_ready(qp, mt * 3 + ks, lane);
      bf16x16 bc = load_frag_ready(Cp, nt * 3 + ks, lane);
      ha = wmma_bf16(aq, bc, ha);
    }
    hacc[q3] = ha;
    Acc nn; nn.v = na;
#pragma unroll
    for (int r = 0; r < 8; ++r) stage[(mt * 16 + r + 8 * hi) * HD + nt * 16 + l15] = nn.f[r];
  }
  __syncthreads();
  if (tid < CSZ) {                                    // denominator per token
    float s = 0.f;
    for (int d = 0; d < HD; ++d) s += (stage[tid * HD + d] + nin[d]) * qf[tid * HD + d];
    den[tid] = fmaxf(s, 1.0f);
  }
  __syncthreads();
#pragma unroll
  for (int q3 = 0; q3 < 3; ++q3) {
    int t = wid * 3 + q3;
    int mt = t / 6, nt = t % 6;
    Acc ha; ha.v = hacc[q3];
#pragma unroll
    for (int r = 0; r < 8; ++r) {
      int i = mt * 16 + r + 8 * hi;
      int d = nt * 16 + l15;
      hbuf[(size_t)(tbase + i) * HIDN + nh * HD + d] = ha.f[r] / den[i];
    }
  }
}

// ---------------- combine: o*h, LN, +skip, *silu(r) -> bf16 ----------------
__global__ void __launch_bounds__(256) combine_kernel(const float* __restrict__ hbuf,
    const float* __restrict__ obuf, const float* __restrict__ xskip, const float* __restrict__ rt,
    const float* __restrict__ lnw, const float* __restrict__ lnb, uint16_t* __restrict__ pre_down) {
  __shared__ float red[256];
  int tok = blockIdx.x, tid = threadIdx.x;
  size_t base = (size_t)tok * HIDN;
  float g[3]; float s = 0.f;
#pragma unroll
  for (int i = 0; i < 3; ++i) {
    int cc = tid + i * 256;
    g[i] = obuf[base + cc] * hbuf[base + cc];
    s += g[i];
  }
  red[tid] = s; __syncthreads();
  for (int o = 128; o > 0; o >>= 1) { if (tid < o) red[tid] += red[tid + o]; __syncthreads(); }
  float mean = red[0] / (float)HIDN; __syncthreads();
  s = 0.f;
#pragma unroll
  for (int i = 0; i < 3; ++i) { float d = g[i] - mean; s += d * d; }
  red[tid] = s; __syncthreads();
  for (int o = 128; o > 0; o >>= 1) { if (tid < o) red[tid] += red[tid + o]; __syncthreads(); }
  float rstd = rsqrtf(red[0] / (float)HIDN + 1e-6f);
#pragma unroll
  for (int i = 0; i < 3; ++i) {
    int cc = tid + i * 256;
    float v = (g[i] - mean) * rstd * lnw[cc] + lnb[cc] + xskip[base + cc];
    float r = rt[base + cc];
    v *= r / (1.0f + __expf(-r));
    pre_down[base + cc] = f2bf(v);
  }
}

// ---------------- host driver ----------------
extern "C" void kernel_launch(void* const* d_in, const int* in_sizes, int n_in,
                              void* d_out, int out_size, void* d_ws, size_t ws_size,
                              hipStream_t stream) {
  const float* x        = (const float*)d_in[0];
  const float* ln_in_w  = (const float*)d_in[1];
  const float* ln_in_b  = (const float*)d_in[2];
  const float* ln_hid_w = (const float*)d_in[3];
  const float* ln_hid_b = (const float*)d_in[4];
  const float* up_l_w   = (const float*)d_in[5];
  const float* up_l_b   = (const float*)d_in[6];
  const float* up_r_w   = (const float*)d_in[7];
  const float* up_r_b   = (const float*)d_in[8];
  const float* down_w   = (const float*)d_in[9];
  const float* down_b   = (const float*)d_in[10];
  const float* conv_w   = (const float*)d_in[11];
  const float* conv_b   = (const float*)d_in[12];
  const float* skip_w   = (const float*)d_in[13];
  const float* skip_b   = (const float*)d_in[14];
  const float* fused_w  = (const float*)d_in[15];
  const float* fused_b  = (const float*)d_in[16];
  const float* wo_w     = (const float*)d_in[17];
  const float* wo_b     = (const float*)d_in[18];
  float* out = (float*)d_out;

  char* wsp = (char*)d_ws;
  size_t off = 0;
  auto alloc = [&](size_t bytes) -> void* {
    void* p = wsp + off;
    off += (bytes + 255) & ~(size_t)255;
    return p;
  };
  uint16_t* xnb    = (uint16_t*)alloc((size_t)NTOK * D_IN * 2);
  float*    xt     = (float*)   alloc((size_t)NTOK * UPD * 4);
  uint16_t* xtb    = (uint16_t*)alloc((size_t)NTOK * UPD * 2);
  uint16_t* xcb    = (uint16_t*)alloc((size_t)NTOK * UPD * 2);
  float*    rt     = (float*)   alloc((size_t)NTOK * HIDN * 4);
  float*    xskip  = (float*)   alloc((size_t)NTOK * HIDN * 4);
  float*    proj   = (float*)   alloc((size_t)NTOK * FUSED * 4);
  float*    obuf   = (float*)   alloc((size_t)NTOK * HIDN * 4);
  float*    hbuf   = (float*)   alloc((size_t)NTOK * HIDN * 4);
  float*    Ccon   = (float*)   alloc((size_t)BATCH * NHH * NC * HD * HD * 4);
  float*    Cini   = (float*)   alloc((size_t)BATCH * NHH * NC * HD * HD * 4);
  float*    ncon   = (float*)   alloc((size_t)BATCH * NHH * NC * HD * 4);
  float*    nini   = (float*)   alloc((size_t)BATCH * NHH * NC * HD * 4);
  uint16_t* pdb    = (uint16_t*)alloc((size_t)NTOK * D_IN * 2);
  uint32_t* upl_p  = (uint32_t*)alloc((size_t)(D_IN / 2) * UPD * 4);
  uint32_t* upr_p  = (uint32_t*)alloc((size_t)(D_IN / 2) * HIDN * 4);
  uint32_t* skip_p = (uint32_t*)alloc((size_t)(UPD / 2) * HIDN * 4);
  uint32_t* fus_p  = (uint32_t*)alloc((size_t)(UPD / 2) * FUSED * 4);
  uint32_t* wo_p   = (uint32_t*)alloc((size_t)(UPD / 2) * HIDN * 4);
  uint32_t* down_p = (uint32_t*)alloc((size_t)(D_IN / 2) * HIDN * 4);
  (void)ws_size; (void)n_in; (void)in_sizes; (void)out_size;

  auto packW = [&](const float* W, uint32_t* Wp, int K, int N) {
    int total = (K / 2) * N;
    pack_weight_kernel<<<(total + 255) / 256, 256, 0, stream>>>(W, Wp, K, N);
  };
  packW(up_l_w,  upl_p,  D_IN, UPD);
  packW(up_r_w,  upr_p,  D_IN, HIDN);
  packW(skip_w,  skip_p, UPD,  HIDN);
  packW(fused_w, fus_p,  UPD,  FUSED);
  packW(wo_w,    wo_p,   UPD,  HIDN);
  packW(down_w,  down_p, D_IN, HIDN);

  ln_in_kernel<<<NTOK, 256, 0, stream>>>(x, ln_in_w, ln_in_b, xnb);

  dim3 blk(256);
  gemm_bf16_kernel<0><<<dim3(UPD / 128, NTOK / 128), blk, 0, stream>>>(xnb, upl_p, up_l_b, nullptr, xt, NTOK, UPD, D_IN);
  gemm_bf16_kernel<0><<<dim3(HIDN / 128, NTOK / 128), blk, 0, stream>>>(xnb, upr_p, up_r_b, nullptr, rt, NTOK, HIDN, D_IN);

  conv_silu_kernel<<<NTOK, 256, 0, stream>>>(xt, conv_w, conv_b, xcb, xtb);

  gemm_bf16_kernel<0><<<dim3(HIDN / 128, NTOK / 128), blk, 0, stream>>>(xcb, skip_p, skip_b, nullptr, xskip, NTOK, HIDN, UPD);
  gemm_bf16_kernel<0><<<dim3((FUSED + 127) / 128, NTOK / 128), blk, 0, stream>>>(xcb, fus_p, fused_b, nullptr, proj, NTOK, FUSED, UPD);
  gemm_bf16_kernel<1><<<dim3(HIDN / 128, NTOK / 128), blk, 0, stream>>>(xtb, wo_p, wo_b, nullptr, obuf, NTOK, HIDN, UPD);

  chunk_contrib_kernel<<<BATCH * NHH * NC, 256, 0, stream>>>(proj, Ccon, ncon);
  scan_kernel<<<BATCH * NHH, 256, 0, stream>>>(Ccon, ncon, Cini, nini);

  const int smem_bytes = SMEM_WORDS * 4;
  hipFuncSetAttribute(reinterpret_cast<const void*>(chunk_out_kernel),
                      hipFuncAttributeMaxDynamicSharedMemorySize, smem_bytes);
  chunk_out_kernel<<<BATCH * NHH * NC, 256, smem_bytes, stream>>>(proj, Cini, nini, hbuf);

  combine_kernel<<<NTOK, 256, 0, stream>>>(hbuf, obuf, xskip, rt, ln_hid_w, ln_hid_b, pdb);

  gemm_bf16_kernel<2><<<dim3(D_IN / 128, NTOK / 128), blk, 0, stream>>>(pdb, down_p, down_b, x, out, NTOK, D_IN, D_IN);
}
